// MultiHeadedAttention_24661702214158
// MI455X (gfx1250) — matrix operand order repaired and checked by reading, not compiled, not run on hardware
//
#include <hip/hip_runtime.h>

typedef __attribute__((ext_vector_type(16))) _Float16 v16h;
typedef __attribute__((ext_vector_type(8)))  float    v8f;
typedef __attribute__((ext_vector_type(4)))  int      vec4i;

#define B_   4
#define S_   2048
#define HID_ 1024
#define H_   16
#define M_   (B_ * S_)   /* 8192 */

#if defined(__HIP_DEVICE_COMPILE__) && \
    __has_builtin(__builtin_amdgcn_global_load_async_to_lds_b128)
#define USE_ASYNC_LDS 1
#else
#define USE_ASYNC_LDS 0
#endif

__device__ __forceinline__ void wait_asynccnt0() {
#if defined(__HIP_DEVICE_COMPILE__) && \
    __has_builtin(__builtin_amdgcn_s_wait_asynccnt)
  __builtin_amdgcn_s_wait_asynccnt(0);
#elif defined(__HIP_DEVICE_COMPILE__)
  asm volatile("s_wait_asynccnt 0x0" ::: "memory");
#endif
}

// ---------------------------------------------------------------------------
// WMMA helper: D = A(16x32 f16) * B(32x16 f16) + C(16x16 f32)
// ---------------------------------------------------------------------------
__device__ __forceinline__ v8f wmma_f16(v16h a, v16h b, v8f c) {
  return __builtin_amdgcn_wmma_f32_16x16x32_f16(false, a, false, b, (short)0, c,
                                                false, false);
}

// A-fragment (16x32, row-major source, stride ld in elements).
// ISA 7.12.2: lanes 0-15 hold M=lane, K={0..7,16..23}; lanes 16-31 hold
// M=lane-16, K={8..15,24..31}.
template <typename T>
__device__ __forceinline__ v16h frag_a(const T* base, int ld) {
  const int lane = threadIdx.x & 31;
  const int m    = lane & 15;
  const int kh   = (lane >> 4) << 3;  // 0 or 8
  const T* row   = base + (size_t)m * ld + kh;
  v16h a;
#pragma unroll
  for (int e = 0; e < 8; ++e) a[e] = (_Float16)row[e];
#pragma unroll
  for (int e = 0; e < 8; ++e) a[8 + e] = (_Float16)row[16 + e];
  return a;
}

// B-fragment (32x16) gathered from a source stored "transposed", i.e. source
// is row-major [N][K] and we need B[k][n] = src[n][k].  Lanes 0-15: N=lane,
// K={0..7,16..23}; lanes 16-31: N=lane-16, K={8..15,24..31}.
template <typename T>
__device__ __forceinline__ v16h frag_bt(const T* base, int ld) {
  const int lane = threadIdx.x & 31;
  const int n    = lane & 15;
  const int kg   = (lane >> 4) << 3;  // 0 or 8
  const T* p     = base + (size_t)n * ld + kg;
  v16h b;
#pragma unroll
  for (int e = 0; e < 8; ++e) b[e] = (_Float16)p[e];
#pragma unroll
  for (int e = 0; e < 8; ++e) b[8 + e] = (_Float16)p[16 + e];
  return b;
}

// ---------------------------------------------------------------------------
// Projection GEMM:  C[M,N] = A[M,K] @ W[N,K]^T + bias   (K = N = 1024, M = 8192)
// MODE 0: C -> f16 [B,H,S,D]      (Q / K heads)
// MODE 1: C -> f16 [B,H,D,S]      (V heads, transposed for PV B-fragments)
// MODE 2: C -> f32 [M,N]          (final output; A is f16, staged with
//                                  GLOBAL_LOAD_ASYNC_TO_LDS when available)
// Block: 256 threads = 8 waves (4x2), tile 128(M) x 128(N); wave tile 32x64
// (8 WMMAs / K-step, A-fragments reused across 4 B-tiles).  K-steps of 32
// staged through LDS with fused fp32->f16 conversion.
// ---------------------------------------------------------------------------
template <int MODE>
__global__ __launch_bounds__(256) void gemm_proj_kernel(const void* __restrict__ Av,
                                                        const float* __restrict__ W,
                                                        const float* __restrict__ bias,
                                                        void* __restrict__ outv) {
  __shared__ _Float16 As[128][40];  // 128x32 tile, stride 40 (16B-aligned rows)
  __shared__ _Float16 Ws[128][40];  // 128x32 tile of W rows (N x K)

  const int tid    = threadIdx.x;
  const int lane   = tid & 31;
  const int wv     = tid >> 5;
  const int wy     = wv >> 1;   // 0..3 (M direction, 32 rows each)
  const int wx     = wv & 1;    // 0..1 (N direction, 64 cols each)
  const int blockN = blockIdx.x * 128;
  const int blockM = blockIdx.y * 128;

  v8f acc[2][4] = {};

  const int ra = tid >> 1, ca = (tid & 1) << 4;  // staging: 16 elems/thread

  for (int k0 = 0; k0 < HID_; k0 += 32) {
    // ---- stage A tile (128 x 32) ----
    if (MODE == 2) {
      const _Float16* src = (const _Float16*)Av +
                            (size_t)(blockM + ra) * HID_ + k0 + ca;
#if USE_ASYNC_LDS
      // Builtin signature (from toolchain diagnostic): first arg is a
      // global-AS pointer to a 128-bit int vector; LDS arg is AS3.
      typedef vec4i __attribute__((address_space(1))) g_vec4i;
      typedef vec4i __attribute__((address_space(3))) l_vec4i;
      _Float16* lp = &As[ra][ca];
      // IOFFSET is added to BOTH the global and LDS address (ISA 10.x), so a
      // single base pair + imm offsets 0/16 moves the 32-byte chunk.
      __builtin_amdgcn_global_load_async_to_lds_b128((g_vec4i*)src,
                                                     (l_vec4i*)lp, 0, 0);
      __builtin_amdgcn_global_load_async_to_lds_b128((g_vec4i*)src,
                                                     (l_vec4i*)lp, 16, 0);
#else
#pragma unroll
      for (int i = 0; i < 16; ++i) As[ra][ca + i] = src[i];
#endif
    } else {
      const float* src = (const float*)Av +
                         (size_t)(blockM + ra) * HID_ + k0 + ca;
#pragma unroll
      for (int i = 0; i < 16; ++i) As[ra][ca + i] = (_Float16)src[i];
    }
    // ---- stage W tile (128 x 32), fp32 -> f16 ----
    {
      const float* src = W + (size_t)(blockN + ra) * HID_ + k0 + ca;
#pragma unroll
      for (int i = 0; i < 16; ++i) Ws[ra][ca + i] = (_Float16)src[i];
    }
    if (MODE == 2) wait_asynccnt0();
    __syncthreads();

    const v16h a0 = frag_a(&As[wy * 32][0], 40);
    const v16h a1 = frag_a(&As[wy * 32 + 16][0], 40);
#pragma unroll
    for (int t = 0; t < 4; ++t) {
      const v16h b = frag_bt(&Ws[wx * 64 + 16 * t][0], 40);
      acc[0][t] = wmma_f16(a0, b, acc[0][t]);
      acc[1][t] = wmma_f16(a1, b, acc[1][t]);
    }
    __syncthreads();
  }

  // ---- epilogue ----
#pragma unroll
  for (int tn = 0; tn < 4; ++tn) {
    const int   n  = blockN + wx * 64 + tn * 16 + (lane & 15);
    const float bn = bias[n];
#pragma unroll
    for (int tm = 0; tm < 2; ++tm) {
#pragma unroll
      for (int r = 0; r < 8; ++r) {
        const int   m   = blockM + wy * 32 + tm * 16 + r + ((lane >> 4) << 3);
        const float val = acc[tm][tn][r] + bn;
        if (MODE == 2) {
          ((float*)outv)[(size_t)m * HID_ + n] = val;
        } else {
          const int bb = m >> 11, s = m & (S_ - 1);
          const int h = n >> 6, d = n & 63;
          if (MODE == 0)
            ((_Float16*)outv)[((size_t)(bb * H_ + h) * S_ + s) * 64 + d] =
                (_Float16)val;
          else
            ((_Float16*)outv)[((size_t)(bb * H_ + h) * 64 + d) * S_ + s] =
                (_Float16)val;
        }
      }
    }
  }
}

// ---------------------------------------------------------------------------
// Flash attention: block = 64 query rows of one (b,h); 4 independent waves,
// each owning a 16-row strip.  No block-level barriers (waves exit the causal
// key loop independently; loop bound is wave-scalar via readfirstlane so EXEC
// stays all-ones around every WMMA).  Softmax runs in the base-2 domain:
// logits are pre-scaled by log2(e)/sqrt(D) so every exponent is one v_exp_f32.
// ---------------------------------------------------------------------------
__global__ __launch_bounds__(128) void attn_kernel(
    const _Float16* __restrict__ Qh, const _Float16* __restrict__ Kh,
    const _Float16* __restrict__ Vt, const int* __restrict__ pad,
    const int* __restrict__ causal_p, _Float16* __restrict__ Hid) {
  __shared__ _Float16 plds[4][16][72];  // per-wave P strip (16x64, stride 72)

  const int bh   = blockIdx.x;          // 0..63 = b*16 + h
  const int bb   = bh >> 4;
  const int h    = bh & 15;
  const int q0   = blockIdx.y << 6;
  const int lane = threadIdx.x & 31;
  const int w    = __builtin_amdgcn_readfirstlane((int)threadIdx.x) >> 5;  // 0..3
  const int qw   = q0 + (w << 4);
  const int causal = causal_p[0];

  const _Float16* Qhead = Qh + (size_t)bh * S_ * 64;
  const _Float16* Khead = Kh + (size_t)bh * S_ * 64;
  const _Float16* Vhead = Vt + (size_t)bh * 64 * S_;

  const v16h aq0 = frag_a(Qhead + (size_t)qw * 64, 64);
  const v16h aq1 = frag_a(Qhead + (size_t)qw * 64 + 32, 64);

  int  qrow[8];
  bool padq[8];
#pragma unroll
  for (int r = 0; r < 8; ++r) {
    qrow[r] = qw + r + ((lane >> 4) << 3);
    padq[r] = (pad[bb * S_ + qrow[r]] == 0);
  }

  float mrow[8], lrow[8];
  v8f   o[4] = {};
#pragma unroll
  for (int r = 0; r < 8; ++r) {
    mrow[r] = -3.0e38f;
    lrow[r] = 0.0f;
  }

  const float SCL = 0.125f * 1.44269504f;  // log2(e) / sqrt(D)

  const int kbEnd = causal ? ((qw + 15) >> 6) : ((S_ >> 6) - 1);
  for (int kb = 0; kb <= kbEnd; ++kb) {
    const int kBase = kb << 6;

    if (kb < kbEnd) {  // scalar-uniform prefetch of next K/V block
      __builtin_prefetch(Khead + (size_t)(kBase + 64) * 64, 0, 3);
      __builtin_prefetch(Vhead + kBase + 64, 0, 3);
    }

    // S = (Q K^T): four 16x16 tiles, contraction over D=64 (2 K-steps)
    v8f sc[4];
#pragma unroll
    for (int j = 0; j < 4; ++j) {
      const _Float16* kp = Khead + (size_t)(kBase + 16 * j) * 64;
      const v16h b0 = frag_bt(kp, 64);
      const v16h b1 = frag_bt(kp + 32, 64);
      v8f z = {};
      sc[j] = wmma_f16(aq0, b0, z);
      sc[j] = wmma_f16(aq1, b1, sc[j]);
    }

    // mask + online softmax (base-2).  C layout: element (r) of tile j in
    // this lane is row = r + 8*(lane>=16), col = 16*j + (lane&15); each row
    // lives in one 16-lane half-group -> butterfly xor reductions 1,2,4,8.
    float corr[8];
#pragma unroll
    for (int r = 0; r < 8; ++r) {
      float mv = -3.0e38f;
#pragma unroll
      for (int j = 0; j < 4; ++j) {
        const int   kcol = kBase + 16 * j + (lane & 15);
        float       vv   = (float)sc[j][r] * SCL;
        if (padq[r]) vv = -1.4426950e9f;              // -1e9 * log2(e)
        if (causal && kcol > qrow[r]) vv = -1.0e30f;  // causal -inf
        sc[j][r] = vv;
        mv       = fmaxf(mv, vv);
      }
#pragma unroll
      for (int off = 1; off < 16; off <<= 1)
        mv = fmaxf(mv, __shfl_xor(mv, off, 32));
      const float mnew = fmaxf(mrow[r], mv);
      corr[r]          = exp2f(mrow[r] - mnew);
      mrow[r]          = mnew;

      float sum = 0.0f;
#pragma unroll
      for (int j = 0; j < 4; ++j) {
        const float p = exp2f(sc[j][r] - mnew);
        sc[j][r]      = p;
        sum += p;
      }
#pragma unroll
      for (int off = 1; off < 16; off <<= 1) sum += __shfl_xor(sum, off, 32);
      lrow[r] = lrow[r] * corr[r] + sum;
    }

#pragma unroll
    for (int jt = 0; jt < 4; ++jt)
#pragma unroll
      for (int r = 0; r < 8; ++r) o[jt][r] *= corr[r];

    // Re-layout P (C-tile regs) -> A-fragment via per-wave LDS strip.
    // Same-wave DS ops are in-order, so no barrier is needed.
#pragma unroll
    for (int j = 0; j < 4; ++j)
#pragma unroll
      for (int r = 0; r < 8; ++r)
        plds[w][r + ((lane >> 4) << 3)][16 * j + (lane & 15)] =
            (_Float16)sc[j][r];

    const v16h ap0 = frag_a(&plds[w][0][0], 72);
    const v16h ap1 = frag_a(&plds[w][0][0] + 32, 72);

    // O += P @ V : contraction over 64 keys (2 steps), V from [D][S] layout.
#pragma unroll
    for (int jt = 0; jt < 4; ++jt) {
      const _Float16* vp = Vhead + (size_t)(16 * jt) * S_ + kBase;
      const v16h bv0 = frag_bt(vp, S_);
      const v16h bv1 = frag_bt(vp + 32, S_);
      o[jt] = wmma_f16(ap0, bv0, o[jt]);
      o[jt] = wmma_f16(ap1, bv1, o[jt]);
    }
  }

#pragma unroll
  for (int r = 0; r < 8; ++r) lrow[r] = 1.0f / lrow[r];
#pragma unroll
  for (int jt = 0; jt < 4; ++jt) {
    const int d = 16 * jt + (lane & 15);
#pragma unroll
    for (int r = 0; r < 8; ++r)
      Hid[(size_t)(bb * S_ + qrow[r]) * HID_ + h * 64 + d] =
          (_Float16)(o[jt][r] * lrow[r]);
  }
}

// ---------------------------------------------------------------------------
// Launch: 3 projections -> fused flash attention -> output projection.
// Workspace: Qh,Kh (f16 [B,H,S,D]), Vt (f16 [B,H,D,S]), Hid (f16 [M,HID])
//            = 4 * 16.78 MB = 67 MB (all L2-resident).
// ---------------------------------------------------------------------------
extern "C" void kernel_launch(void* const* d_in, const int* in_sizes, int n_in,
                              void* d_out, int out_size, void* d_ws,
                              size_t ws_size, hipStream_t stream) {
  (void)in_sizes; (void)n_in; (void)out_size; (void)ws_size;

  const float* q   = (const float*)d_in[0];
  const float* k   = (const float*)d_in[1];
  const float* v   = (const float*)d_in[2];
  const int*   pad = (const int*)d_in[3];
  const float* Wq  = (const float*)d_in[4];
  const float* bq  = (const float*)d_in[5];
  const float* Wk  = (const float*)d_in[6];
  const float* bk  = (const float*)d_in[7];
  const float* Wv  = (const float*)d_in[8];
  const float* bv  = (const float*)d_in[9];
  const float* Wo  = (const float*)d_in[10];
  const float* bo  = (const float*)d_in[11];
  const int*   isc = (const int*)d_in[12];

  const size_t n16 = (size_t)M_ * HID_;  // 8,388,608 elements per f16 tensor
  _Float16* Qh  = (_Float16*)d_ws;
  _Float16* Kh  = Qh + n16;
  _Float16* Vt  = Kh + n16;
  _Float16* Hid = Vt + n16;

  const dim3 gb(HID_ / 128, M_ / 128);
  const dim3 tb(256);

  gemm_proj_kernel<0><<<gb, tb, 0, stream>>>((const void*)q, Wq, bq, (void*)Qh);
  gemm_proj_kernel<0><<<gb, tb, 0, stream>>>((const void*)k, Wk, bk, (void*)Kh);
  gemm_proj_kernel<1><<<gb, tb, 0, stream>>>((const void*)v, Wv, bv, (void*)Vt);

  attn_kernel<<<dim3(B_ * H_, S_ / 64), dim3(128), 0, stream>>>(Qh, Kh, Vt, pad,
                                                                isc, Hid);

  gemm_proj_kernel<2><<<gb, tb, 0, stream>>>((const void*)Hid, Wo, bo, d_out);
}
